// minGRULM_15977278341607
// MI455X (gfx1250) — matrix operand order repaired and checked by reading, not compiled, not run on hardware
//
#include <hip/hip_runtime.h>
#include <hip/hip_bf16.h>
#include <math.h>

// ---------------------------------------------------------------------------
// minGRU LM forward for MI455X (gfx1250, wave32, WMMA bf16->f32)
// ---------------------------------------------------------------------------

#define S_LEN 8192
#define D_DIM 768
#define DI_DIM 1152
#define L_NUM 6
#define V_DIM 32000
#define FF_DIM 3072
#define CHUNK 128
#define NCHUNK (S_LEN / CHUNK)   // 64

typedef __attribute__((ext_vector_type(16))) __bf16 v16bf;
typedef __attribute__((ext_vector_type(8)))  float  v8f;

union Frag {
    uint4  u[2];   // 32 bytes
    v16bf  bf;
};

__device__ __forceinline__ unsigned short f2bf(float f) {
    union { float f; unsigned u; } v; v.f = f;
    unsigned r = v.u + 0x7FFFu + ((v.u >> 16) & 1u);  // round-to-nearest-even
    return (unsigned short)(r >> 16);
}

// ---------------------------------------------------------------------------
// Weight convert + transpose: W[K,N] fp32 row-major  ->  Wt[N,K] bf16
// ---------------------------------------------------------------------------
__global__ void convT_kernel(const float* __restrict__ W,
                             unsigned short* __restrict__ Wt,
                             int K, int N) {
    int idx = blockIdx.x * blockDim.x + threadIdx.x;
    if (idx >= K * N) return;
    int k = idx / N;
    int n = idx - k * N;
    Wt[(size_t)n * K + k] = f2bf(W[idx]);
}

// ---------------------------------------------------------------------------
// Embedding gather: x[s,d] = emb[tokens[s], d]
// ---------------------------------------------------------------------------
__global__ void embed_kernel(const int* __restrict__ tokens,
                             const float* __restrict__ emb,
                             float* __restrict__ x) {
    int i = blockIdx.x * blockDim.x + threadIdx.x;
    if (i >= S_LEN * D_DIM) return;
    int s = i / D_DIM;
    int d = i - s * D_DIM;
    x[i] = emb[(size_t)tokens[s] * D_DIM + d];
}

// ---------------------------------------------------------------------------
// RMSNorm (F.normalize * sqrt(D) * (gamma+1)) -> bf16 output
// one block (256 threads) per row
// ---------------------------------------------------------------------------
__global__ __launch_bounds__(256)
void rmsnorm_kernel(const float* __restrict__ x,
                    const float* __restrict__ gamma,
                    unsigned short* __restrict__ out) {
    __shared__ float red[256];
    const int row = blockIdx.x;
    const float* xr = x + (size_t)row * D_DIM;
    float s = 0.f;
    for (int d = threadIdx.x; d < D_DIM; d += 256) {
        float v = xr[d];
        s += v * v;
    }
    red[threadIdx.x] = s;
    __syncthreads();
    for (int w = 128; w > 0; w >>= 1) {
        if (threadIdx.x < w) red[threadIdx.x] += red[threadIdx.x + w];
        __syncthreads();
    }
    float norm = fmaxf(sqrtf(red[0]), 1e-12f);
    float sc = 27.712812921102035f / norm;  // sqrt(768)/||x||
    for (int d = threadIdx.x; d < D_DIM; d += 256)
        out[(size_t)row * D_DIM + d] = f2bf(xr[d] * sc * (gamma[d] + 1.f));
}

// ---------------------------------------------------------------------------
// WMMA GEMM:  C[M,N] = A[M,K](bf16,row-major) * Bt[N,K](bf16, i.e. B^T)
// One wave computes 32(M)x64(N): 2 A-frags x 4 B-frags -> 8 WMMAs per K-step.
// Software pipeline with two STATICALLY-NAMED fragment buffers (no runtime
// array indexing -> no scratch spills); K loop manually unrolled 2x.
// Requires K % 64 == 0 (true for 768 / 1152 / 3072).
// Block = 8 M-stacked waves = 256(M)x64(N); grid = (N/64, M/256).
// EPI: 0 = plain fp32 store (RT)
//      1 = +bias, exact GELU, store bf16 (Cbf)
//      2 = +bias, +residual, store fp32
//      3 = +residual, store fp32
//      4 = plain fp32 store, non-temporal (streaming output, e.g. logits)
// ---------------------------------------------------------------------------

// Load one K-step (32 wide) of fragments into 6 named Frag variables.
#define LOAD_STEP(k0, A0, A1, B0, B1, B2, B3)                                  \
    do {                                                                       \
        const unsigned short* ap0_ = arow0 + (k0) + kb * 8;                    \
        A0.u[0] = *(const uint4*)(ap0_);                                       \
        A0.u[1] = *(const uint4*)(ap0_ + 16);                                  \
        const unsigned short* ap1_ = arow1 + (k0) + kb * 8;                    \
        A1.u[0] = *(const uint4*)(ap1_);                                       \
        A1.u[1] = *(const uint4*)(ap1_ + 16);                                  \
        const unsigned short* bp_ = bcol + (k0) + kb * 8;                      \
        B0.u[0] = *(const uint4*)(bp_);                                        \
        B0.u[1] = *(const uint4*)(bp_ + 16);                                   \
        B1.u[0] = *(const uint4*)(bp_ + (size_t)16 * K);                       \
        B1.u[1] = *(const uint4*)(bp_ + (size_t)16 * K + 16);                  \
        B2.u[0] = *(const uint4*)(bp_ + (size_t)32 * K);                       \
        B2.u[1] = *(const uint4*)(bp_ + (size_t)32 * K + 16);                  \
        B3.u[0] = *(const uint4*)(bp_ + (size_t)48 * K);                       \
        B3.u[1] = *(const uint4*)(bp_ + (size_t)48 * K + 16);                  \
    } while (0)

#define WMMA_STEP(A0, A1, B0, B1, B2, B3)                                      \
    do {                                                                       \
        acc[0][0] = __builtin_amdgcn_wmma_f32_16x16x32_bf16(                   \
            false, A0.bf, false, B0.bf, (short)0, acc[0][0], false, false);    \
        acc[0][1] = __builtin_amdgcn_wmma_f32_16x16x32_bf16(                   \
            false, A0.bf, false, B1.bf, (short)0, acc[0][1], false, false);    \
        acc[0][2] = __builtin_amdgcn_wmma_f32_16x16x32_bf16(                   \
            false, A0.bf, false, B2.bf, (short)0, acc[0][2], false, false);    \
        acc[0][3] = __builtin_amdgcn_wmma_f32_16x16x32_bf16(                   \
            false, A0.bf, false, B3.bf, (short)0, acc[0][3], false, false);    \
        acc[1][0] = __builtin_amdgcn_wmma_f32_16x16x32_bf16(                   \
            false, A1.bf, false, B0.bf, (short)0, acc[1][0], false, false);    \
        acc[1][1] = __builtin_amdgcn_wmma_f32_16x16x32_bf16(                   \
            false, A1.bf, false, B1.bf, (short)0, acc[1][1], false, false);    \
        acc[1][2] = __builtin_amdgcn_wmma_f32_16x16x32_bf16(                   \
            false, A1.bf, false, B2.bf, (short)0, acc[1][2], false, false);    \
        acc[1][3] = __builtin_amdgcn_wmma_f32_16x16x32_bf16(                   \
            false, A1.bf, false, B3.bf, (short)0, acc[1][3], false, false);    \
    } while (0)

template <int EPI>
__global__ __launch_bounds__(256)
void wmma_gemm_kernel(const unsigned short* __restrict__ A,
                      const unsigned short* __restrict__ Bt,
                      float* __restrict__ C,
                      unsigned short* __restrict__ Cbf,
                      const float* __restrict__ bias,
                      const float* __restrict__ resid,
                      int M, int N, int K) {
    const int lane = threadIdx.x & 31;
    const int wave = threadIdx.x >> 5;
    const int n0 = blockIdx.x * 64;
    const int m0 = (blockIdx.y * 8 + wave) * 32;
    const int l15 = lane & 15;
    const int kb  = lane >> 4;           // 0 or 1 (lane half)

    v8f acc[2][4];
#pragma unroll
    for (int r = 0; r < 2; ++r)
#pragma unroll
        for (int t = 0; t < 4; ++t)
            acc[r][t] = (v8f){0.f,0.f,0.f,0.f,0.f,0.f,0.f,0.f};

    const unsigned short* arow0 = A + (size_t)(m0 + l15) * K;
    const unsigned short* arow1 = A + (size_t)(m0 + 16 + l15) * K;
    const unsigned short* bcol  = Bt + (size_t)(n0 + l15) * K;

    // ping buffer (x*) and pong buffer (y*): statically named, never
    // indexed dynamically -> stays in VGPRs.
    Frag xa0, xa1, xb0, xb1, xb2, xb3;
    Frag ya0, ya1, yb0, yb1, yb2, yb3;

    LOAD_STEP(0, xa0, xa1, xb0, xb1, xb2, xb3);
    for (int k0 = 0; k0 < K; k0 += 64) {
        LOAD_STEP(k0 + 32, ya0, ya1, yb0, yb1, yb2, yb3);  // overlaps WMMAs
        WMMA_STEP(xa0, xa1, xb0, xb1, xb2, xb3);
        if (k0 + 64 < K)
            LOAD_STEP(k0 + 64, xa0, xa1, xb0, xb1, xb2, xb3);  // overlaps WMMAs
        WMMA_STEP(ya0, ya1, yb0, yb1, yb2, yb3);
    }

    // C/D layout: lanes 0-15 -> M=i, lanes 16-31 -> M=i+8; N = lane&15
#pragma unroll
    for (int r = 0; r < 2; ++r) {
        const int mbase = m0 + r * 16 + kb * 8;
#pragma unroll
        for (int t = 0; t < 4; ++t) {
            const int n = n0 + t * 16 + l15;
#pragma unroll
            for (int i = 0; i < 8; ++i) {
                const int m = mbase + i;
                const size_t off = (size_t)m * N + n;
                float v = acc[r][t][i];
                if (EPI == 1) {
                    v += bias[n];
                    v = 0.5f * v * (1.f + erff(v * 0.70710678118654752f));
                    Cbf[off] = f2bf(v);
                } else if (EPI == 4) {
                    __builtin_nontemporal_store(v, &C[off]);
                } else {
                    if (EPI == 2) v += bias[n];
                    if (EPI == 2 || EPI == 3) v += resid[off];
                    C[off] = v;
                }
            }
        }
    }
}

// ---------------------------------------------------------------------------
// minGRU scan, pass 1: per (chunk, channel) local scan.
// hg[S, 2*DI]: hidden = [:, :DI], gate = [:, DI:]
// h_t = (1-z)h_{t-1} + z*g(h~),  z = sigmoid(gate),
// g(x) = x>=0 ? x+0.5 : sigmoid(x)
// ---------------------------------------------------------------------------
__global__ void scan_pass1_kernel(const float* __restrict__ hg,
                                  float* __restrict__ hloc,
                                  float* __restrict__ apre,
                                  float* __restrict__ aprod,
                                  float* __restrict__ hend) {
    int idx = blockIdx.x * blockDim.x + threadIdx.x;
    if (idx >= NCHUNK * DI_DIM) return;
    const int j = idx / DI_DIM;        // chunk
    const int c = idx - j * DI_DIM;    // channel
    float h = 0.f, ar = 1.f;
    for (int i = 0; i < CHUNK; ++i) {
        const int s = j * CHUNK + i;
        const float hid = hg[(size_t)s * (2 * DI_DIM) + c];
        const float gt  = hg[(size_t)s * (2 * DI_DIM) + DI_DIM + c];
        const float z = 1.f / (1.f + expf(-gt));
        const float g = (hid >= 0.f) ? (hid + 0.5f) : 1.f / (1.f + expf(-hid));
        const float a = 1.f - z;
        h = a * h + z * g;
        ar *= a;
        hloc[(size_t)s * DI_DIM + c] = h;
        apre[(size_t)s * DI_DIM + c] = ar;
    }
    aprod[(size_t)j * DI_DIM + c] = ar;
    hend [(size_t)j * DI_DIM + c] = h;
}

// pass 2: per-channel carry scan across chunk summaries
__global__ void scan_pass2_kernel(const float* __restrict__ aprod,
                                  const float* __restrict__ hend,
                                  float* __restrict__ carry) {
    int c = blockIdx.x * blockDim.x + threadIdx.x;
    if (c >= DI_DIM) return;
    float cr = 0.f;
    for (int j = 0; j < NCHUNK; ++j) {
        carry[(size_t)j * DI_DIM + c] = cr;
        cr = aprod[(size_t)j * DI_DIM + c] * cr + hend[(size_t)j * DI_DIM + c];
    }
}

// pass 3: fixup h = hloc + apre * carry[chunk] -> bf16
__global__ void scan_pass3_kernel(const float* __restrict__ hloc,
                                  const float* __restrict__ apre,
                                  const float* __restrict__ carry,
                                  unsigned short* __restrict__ hbf) {
    int i = blockIdx.x * blockDim.x + threadIdx.x;
    if (i >= S_LEN * DI_DIM) return;
    const int s = i / DI_DIM;
    const int c = i - s * DI_DIM;
    const int j = s / CHUNK;
    hbf[i] = f2bf(hloc[i] + apre[i] * carry[(size_t)j * DI_DIM + c]);
}

// ---------------------------------------------------------------------------
// Orchestration
// ---------------------------------------------------------------------------
extern "C" void kernel_launch(void* const* d_in, const int* in_sizes, int n_in,
                              void* d_out, int out_size, void* d_ws, size_t ws_size,
                              hipStream_t stream) {
    (void)in_sizes; (void)n_in; (void)out_size; (void)ws_size;

    const int*   tokens   = (const int*)  d_in[0];
    const float* emb      = (const float*)d_in[1];
    const float* gamma1   = (const float*)d_in[2];
    const float* W_hg     = (const float*)d_in[3];   // [L, D, 2*DI]
    const float* W_out    = (const float*)d_in[4];   // [L, DI, D]
    const float* gamma2   = (const float*)d_in[5];
    const float* W_ff1    = (const float*)d_in[6];   // [L, D, 4D]
    const float* b_ff1    = (const float*)d_in[7];   // [L, 4D]
    const float* W_ff2    = (const float*)d_in[8];   // [L, 4D, D]
    const float* b_ff2    = (const float*)d_in[9];   // [L, D]
    const float* gamma_f  = (const float*)d_in[10];
    const float* W_logits = (const float*)d_in[11];  // [D, V]
    float* out = (float*)d_out;                       // [S, V] fp32

    // ---- workspace carve-up (256B aligned blocks) ----
    char* ws = (char*)d_ws;
    size_t off = 0;
    auto carve = [&](size_t bytes) -> char* {
        char* p = ws + off;
        off += (bytes + 255) & ~(size_t)255;
        return p;
    };
    float*          x     = (float*)         carve((size_t)S_LEN * D_DIM * 4);
    unsigned short* xn    = (unsigned short*)carve((size_t)S_LEN * D_DIM * 2);
    float*          hg    = (float*)         carve((size_t)S_LEN * 2 * DI_DIM * 4);
    float*          hloc  = (float*)         carve((size_t)S_LEN * DI_DIM * 4);
    float*          apre  = (float*)         carve((size_t)S_LEN * DI_DIM * 4);
    float*          aprod = (float*)         carve((size_t)NCHUNK * DI_DIM * 4);
    float*          hend  = (float*)         carve((size_t)NCHUNK * DI_DIM * 4);
    float*          carry = (float*)         carve((size_t)NCHUNK * DI_DIM * 4);
    unsigned short* hbf   = (unsigned short*)carve((size_t)S_LEN * DI_DIM * 2);
    unsigned short* ffbf  = (unsigned short*)carve((size_t)S_LEN * FF_DIM * 2);
    unsigned short* WhgT  = (unsigned short*)carve((size_t)L_NUM * 2 * DI_DIM * D_DIM * 2);
    unsigned short* WoutT = (unsigned short*)carve((size_t)L_NUM * D_DIM * DI_DIM * 2);
    unsigned short* Wff1T = (unsigned short*)carve((size_t)L_NUM * FF_DIM * D_DIM * 2);
    unsigned short* Wff2T = (unsigned short*)carve((size_t)L_NUM * D_DIM * FF_DIM * 2);
    unsigned short* WlogT = (unsigned short*)carve((size_t)V_DIM * D_DIM * 2);

    const int TB = 256;
    auto cdiv = [](size_t a, int b) { return (unsigned)((a + b - 1) / b); };

    // ---- one-shot (per call) bf16 weight conversion + transpose ----
    for (int l = 0; l < L_NUM; ++l) {
        convT_kernel<<<cdiv((size_t)D_DIM * 2 * DI_DIM, TB), TB, 0, stream>>>(
            W_hg + (size_t)l * D_DIM * 2 * DI_DIM,
            WhgT + (size_t)l * 2 * DI_DIM * D_DIM, D_DIM, 2 * DI_DIM);
        convT_kernel<<<cdiv((size_t)DI_DIM * D_DIM, TB), TB, 0, stream>>>(
            W_out + (size_t)l * DI_DIM * D_DIM,
            WoutT + (size_t)l * D_DIM * DI_DIM, DI_DIM, D_DIM);
        convT_kernel<<<cdiv((size_t)D_DIM * FF_DIM, TB), TB, 0, stream>>>(
            W_ff1 + (size_t)l * D_DIM * FF_DIM,
            Wff1T + (size_t)l * FF_DIM * D_DIM, D_DIM, FF_DIM);
        convT_kernel<<<cdiv((size_t)FF_DIM * D_DIM, TB), TB, 0, stream>>>(
            W_ff2 + (size_t)l * FF_DIM * D_DIM,
            Wff2T + (size_t)l * D_DIM * FF_DIM, FF_DIM, D_DIM);
    }
    convT_kernel<<<cdiv((size_t)D_DIM * V_DIM, TB), TB, 0, stream>>>(
        W_logits, WlogT, D_DIM, V_DIM);

    // ---- embedding ----
    embed_kernel<<<cdiv((size_t)S_LEN * D_DIM, TB), TB, 0, stream>>>(tokens, emb, x);

    const dim3 blk(256);
    for (int l = 0; l < L_NUM; ++l) {
        // x_n = rmsnorm(x, gamma1[l])
        rmsnorm_kernel<<<S_LEN, blk, 0, stream>>>(x, gamma1 + (size_t)l * D_DIM, xn);

        // hg = x_n @ W_hg[l]   [S, 2*DI]
        wmma_gemm_kernel<0><<<dim3(2 * DI_DIM / 64, S_LEN / 256), blk, 0, stream>>>(
            xn, WhgT + (size_t)l * 2 * DI_DIM * D_DIM, hg, nullptr, nullptr, nullptr,
            S_LEN, 2 * DI_DIM, D_DIM);

        // chunked minGRU scan -> hbf (bf16)
        scan_pass1_kernel<<<cdiv((size_t)NCHUNK * DI_DIM, TB), TB, 0, stream>>>(
            hg, hloc, apre, aprod, hend);
        scan_pass2_kernel<<<cdiv((size_t)DI_DIM, TB), TB, 0, stream>>>(
            aprod, hend, carry);
        scan_pass3_kernel<<<cdiv((size_t)S_LEN * DI_DIM, TB), TB, 0, stream>>>(
            hloc, apre, carry, hbf);

        // x = h @ W_out[l] + x
        wmma_gemm_kernel<3><<<dim3(D_DIM / 64, S_LEN / 256), blk, 0, stream>>>(
            hbf, WoutT + (size_t)l * D_DIM * DI_DIM, x, nullptr, nullptr, x,
            S_LEN, D_DIM, DI_DIM);

        // h_n = rmsnorm(x, gamma2[l])
        rmsnorm_kernel<<<S_LEN, blk, 0, stream>>>(x, gamma2 + (size_t)l * D_DIM, xn);

        // ff = gelu(h_n @ W_ff1[l] + b_ff1[l])  (bf16)
        wmma_gemm_kernel<1><<<dim3(FF_DIM / 64, S_LEN / 256), blk, 0, stream>>>(
            xn, Wff1T + (size_t)l * FF_DIM * D_DIM, nullptr, ffbf,
            b_ff1 + (size_t)l * FF_DIM, nullptr, S_LEN, FF_DIM, D_DIM);

        // x = ff @ W_ff2[l] + b_ff2[l] + x
        wmma_gemm_kernel<2><<<dim3(D_DIM / 64, S_LEN / 256), blk, 0, stream>>>(
            ffbf, Wff2T + (size_t)l * D_DIM * FF_DIM, x, nullptr,
            b_ff2 + (size_t)l * D_DIM, x, S_LEN, D_DIM, FF_DIM);
    }

    // logits = rmsnorm(x, gamma_f) @ W_logits  (streaming NT store)
    rmsnorm_kernel<<<S_LEN, blk, 0, stream>>>(x, gamma_f, xn);
    wmma_gemm_kernel<4><<<dim3(V_DIM / 64, S_LEN / 256), blk, 0, stream>>>(
        xn, WlogT, out, nullptr, nullptr, nullptr, S_LEN, V_DIM, D_DIM);
}
